// RQKmeans_60172491817566
// MI455X (gfx1250) — matrix operand order repaired and checked by reading, not compiled, not run on hardware
//
#include <hip/hip_runtime.h>

// RQ-Kmeans codebook assignment for MI455X (gfx1250, wave32).
//   N=131072 points, D=128, 4 books x 1024 codes.
//   dist = ||c||^2 - 2 x.c   (||x||^2 dropped: constant per row)
// Pre-pass converts codebooks to bf16 once (1 MB, L2-resident). Each block
// stages the current book's full bf16 codebook into LDS (async global->LDS,
// 272-B padded pitch => conflict-free ds_load_b128). Hot loop is software-
// pipelined (double-buffered tiles): ds_load_b128 for tile ct+1 overlap the
// 8 V_WMMA_F32_16X16X32_BF16 of tile ct, plus dual-issued argmin bookkeeping.

#define N_PTS  131072
#define DIMS   128
#define NBOOK  4
#define KCODES 1024
#define NTILE  (KCODES / 16)                 // 64 code-tiles per book

#define LDS_PITCH   272                      // 256 B row + 16 B pad (4-bank shift)
#define LDS_CB_OFF  4096                     // csq occupies [0, 4KB)
#define LDS_BYTES   (LDS_CB_OFF + KCODES * LDS_PITCH)   // ~276 KB of 320 KB

typedef __attribute__((ext_vector_type(16))) __bf16        v16bf;
typedef __attribute__((ext_vector_type(8)))  float         v8f;
typedef __attribute__((ext_vector_type(8)))  unsigned int  v8u;
typedef __attribute__((ext_vector_type(4)))  int           v4i;

#if defined(__has_builtin)
#if __has_builtin(__builtin_amdgcn_global_load_async_to_lds_b128) && \
    __has_builtin(__builtin_amdgcn_s_wait_asynccnt)
#define HAVE_ASYNC_LDS 1
#endif
#endif
#ifndef HAVE_ASYNC_LDS
#define HAVE_ASYNC_LDS 0
#endif

// pointer-to-addrspace-qualified-int4 types for the async builtin
typedef __attribute__((address_space(1))) v4i* g_v4i_p;   // global
typedef __attribute__((address_space(3))) v4i* l_v4i_p;   // LDS

static __device__ __forceinline__ unsigned int f2bf(float f) {
    union { float f; unsigned int u; } cv; cv.f = f;
    unsigned int u = cv.u;
    return (u + 0x7FFFu + ((u >> 16) & 1u)) >> 16;   // round-to-nearest-even
}
static __device__ __forceinline__ unsigned int pk2(float lo, float hi) {
    return f2bf(lo) | (f2bf(hi) << 16);
}
static __device__ __forceinline__ v16bf pack16(float4 a, float4 b, float4 c, float4 d) {
    v8u p;
    p[0] = pk2(a.x, a.y); p[1] = pk2(a.z, a.w);
    p[2] = pk2(b.x, b.y); p[3] = pk2(b.z, b.w);
    p[4] = pk2(c.x, c.y); p[5] = pk2(c.z, c.w);
    p[6] = pk2(d.x, d.y); p[7] = pk2(d.z, d.w);
    return __builtin_bit_cast(v16bf, p);
}
static __device__ __forceinline__ v16bf ld16bf_a16(const char* p) {  // 16-B aligned 32-B load
    uint4 q0 = *(const uint4*)p;
    uint4 q1 = *(const uint4*)(p + 16);
    v8u u;
    u[0] = q0.x; u[1] = q0.y; u[2] = q0.z; u[3] = q0.w;
    u[4] = q1.x; u[5] = q1.y; u[6] = q1.z; u[7] = q1.w;
    return __builtin_bit_cast(v16bf, u);
}

// ---- pre-pass 1: csq[b*K + k] = sum_d C[b][k][d]^2 (exact fp32) ----
__global__ __launch_bounds__(256) void rqk_csq_kernel(const float* __restrict__ cb,
                                                      float* __restrict__ csq) {
    const int t = blockIdx.x * 256 + threadIdx.x;       // 0 .. 4095
    const float4* p = (const float4*)(cb + (size_t)t * DIMS);
    float s = 0.0f;
#pragma unroll
    for (int i = 0; i < DIMS / 4; ++i) {
        float4 v = p[i];
        s += v.x * v.x + v.y * v.y + v.z * v.z + v.w * v.w;
    }
    csq[t] = s;
}

// ---- pre-pass 2: convert codebooks fp32 -> bf16 once (row-major kept) ----
__global__ __launch_bounds__(256) void rqk_cvt_kernel(const float* __restrict__ cb,
                                                      unsigned int* __restrict__ cbh) {
    const int t = blockIdx.x * 256 + threadIdx.x;       // 4 floats / thread
    float4 v = ((const float4*)cb)[t];
    cbh[2 * t + 0] = pk2(v.x, v.y);
    cbh[2 * t + 1] = pk2(v.z, v.w);
}

struct TileB { v16bf f[4]; float cs; };

// ---- main: one wave = 32 rows (two 16x16 M-tiles), streams 64 code-tiles ----
__global__ __launch_bounds__(256) void rqk_assign_kernel(const float* __restrict__ x,
                                                         const __bf16* __restrict__ cbh,
                                                         const float* __restrict__ csq_g,
                                                         int* __restrict__ out) {
    extern __shared__ char smem[];                      // dynamic: ~276 KB of 320 KB WGP LDS
    float* s_csq = (float*)smem;
    char*  s_cb  = smem + LDS_CB_OFF;                   // [1024 rows][272 B pitch]

    const int lane    = threadIdx.x & 31;               // wave32
    const int wave    = threadIdx.x >> 5;               // 8 waves / block
    const int r       = lane & 15;
    const int hi      = lane >> 4;                      // 0 or 1
    const int rowBase = blockIdx.x * 256 + wave * 32;

    // A fragments: 32 rows x 128 dims as bf16, converted ONCE, live in VGPRs.
    // A layout: lane l<16 -> row M=l, dims K0..15; l>=16 -> dims K16..31 (matches B)
    v16bf a[2][4];
#pragma unroll
    for (int m = 0; m < 2; ++m) {
#pragma unroll
        for (int kf = 0; kf < 4; ++kf) {
            const float* ap = x + (size_t)(rowBase + m * 16 + r) * DIMS + kf * 32 + hi * 16;
            float4 f0 = ((const float4*)ap)[0];
            float4 f1 = ((const float4*)ap)[1];
            float4 f2 = ((const float4*)(ap + 8))[0];
            float4 f3 = ((const float4*)(ap + 8))[1];
            a[m][kf] = pack16(f0, f1, f2, f3);
        }
    }

    for (int b = 0; b < NBOOK; ++b) {
        // ---- stage ||c||^2 and the whole bf16 codebook for book b into LDS ----
        for (int i = threadIdx.x; i < KCODES; i += 256)
            s_csq[i] = csq_g[b * KCODES + i];

        const __bf16* gsrc_base = cbh + (size_t)b * KCODES * DIMS;
        if (b + 1 < NBOOK)                               // warm L2 for the next book
            __builtin_prefetch(gsrc_base + (size_t)KCODES * DIMS, 0, 0);
        // 1024 rows x 256 B, copied as 16384 x 16-B chunks, 64 per thread
        for (int c = threadIdx.x; c < KCODES * 16; c += 256) {
            const int row = c >> 4, seg = c & 15;
            const __bf16* gp = gsrc_base + row * DIMS + seg * 8;
            char*         lp = s_cb + row * LDS_PITCH + seg * 16;
#if HAVE_ASYNC_LDS
            __builtin_amdgcn_global_load_async_to_lds_b128(
                (g_v4i_p)(const void*)gp, (l_v4i_p)(void*)lp, 0, 0);
#else
            *(uint4*)lp = *(const uint4*)gp;
#endif
        }
#if HAVE_ASYNC_LDS
        __builtin_amdgcn_s_wait_asynccnt(0);
#endif
        __syncthreads();

        float minv[2][8];
        int   mini[2][8];
#pragma unroll
        for (int m = 0; m < 2; ++m)
#pragma unroll
            for (int v = 0; v < 8; ++v) { minv[m][v] = 3.0e38f; mini[m][v] = 0; }

        // B layout: lane l<16 -> code column N=l, dims K0..15; l>=16 -> dims K16..31.
        // LDS pitch 272 B = 4-bank shift/row -> the 16 lanes' b128 reads are conflict-free.
        const char* bbase = s_cb + r * LDS_PITCH + hi * 32;

        auto load_tile = [&](int ct) -> TileB {
            TileB t;
            const char* bp = bbase + ct * 16 * LDS_PITCH;
#pragma unroll
            for (int kf = 0; kf < 4; ++kf) t.f[kf] = ld16bf_a16(bp + kf * 64);
            t.cs = s_csq[ct * 16 + r];
            return t;
        };
        auto compute = [&](const TileB& t, int ct) {
            v8f acc0 = {}, acc1 = {};
#pragma unroll
            for (int kf = 0; kf < 4; ++kf) {
                acc0 = __builtin_amdgcn_wmma_f32_16x16x32_bf16(
                    false, a[0][kf], false, t.f[kf], (short)0, acc0, false, false);
                acc1 = __builtin_amdgcn_wmma_f32_16x16x32_bf16(
                    false, a[1][kf], false, t.f[kf], (short)0, acc1, false, false);
            }
            const int code = ct * 16 + r;               // this lane's code column
#pragma unroll
            for (int v = 0; v < 8; ++v) {
                float d0 = fmaf(-2.0f, acc0[v], t.cs);
                if (d0 < minv[0][v]) { minv[0][v] = d0; mini[0][v] = code; }
                float d1 = fmaf(-2.0f, acc1[v], t.cs);
                if (d1 < minv[1][v]) { minv[1][v] = d1; mini[1][v] = code; }
            }
        };

        // software pipeline: loads of tile ct+1 overlap WMMAs of tile ct
        TileB tA = load_tile(0);
        for (int ct = 0; ct < NTILE; ct += 2) {
            TileB tB = load_tile(ct + 1);               // ct+1 <= 63 always
            compute(tA, ct);
            if (ct + 2 < NTILE) tA = load_tile(ct + 2);
            compute(tB, ct + 1);
        }

        // butterfly min+argmin across the 16 lanes sharing each row
        // (first-min tie-break on lowest index, matching jnp.argmin)
#pragma unroll
        for (int off = 1; off < 16; off <<= 1) {
#pragma unroll
            for (int m = 0; m < 2; ++m)
#pragma unroll
                for (int v = 0; v < 8; ++v) {
                    float ov = __shfl_xor(minv[m][v], off, 32);
                    int   oi = __shfl_xor(mini[m][v], off, 32);
                    if (ov < minv[m][v] || (ov == minv[m][v] && oi < mini[m][v])) {
                        minv[m][v] = ov; mini[m][v] = oi;
                    }
                }
        }

        if (r == 0) {   // lanes 0 and 16 hold rows v and v+8 of each M-tile
#pragma unroll
            for (int m = 0; m < 2; ++m)
#pragma unroll
                for (int v = 0; v < 8; ++v) {
                    int row = rowBase + m * 16 + hi * 8 + v;
                    out[(size_t)row * NBOOK + b] = mini[m][v];
                }
        }
        __syncthreads();    // protect LDS before next book's staging
    }
}

extern "C" void kernel_launch(void* const* d_in, const int* in_sizes, int n_in,
                              void* d_out, int out_size, void* d_ws, size_t ws_size,
                              hipStream_t stream) {
    (void)in_sizes; (void)n_in; (void)out_size; (void)ws_size;
    const float* x   = (const float*)d_in[0];   // [N, 128] fp32
    const float* cb  = (const float*)d_in[1];   // [4, 1024, 128] fp32

    // workspace layout: [0, 16KB) csq fp32 ; [16KB, 16KB+1MB) codebooks bf16
    float*        csq = (float*)d_ws;
    unsigned int* cbh = (unsigned int*)((char*)d_ws + 16 * 1024);
    int*          out = (int*)d_out;            // [N, 4] int32

    rqk_csq_kernel<<<dim3((NBOOK * KCODES) / 256), dim3(256), 0, stream>>>(cb, csq);
    // 4*1024*128 = 524288 floats, 4 per thread -> 131072 threads
    rqk_cvt_kernel<<<dim3(512), dim3(256), 0, stream>>>(cb, cbh);
    rqk_assign_kernel<<<dim3(N_PTS / 256), dim3(256), LDS_BYTES, stream>>>(
        x, (const __bf16*)cbh, csq, out);
}
